// LocalAttention_10462540333451
// MI455X (gfx1250) — matrix operand order repaired and checked by reading, not compiled, not run on hardware
//
#include <hip/hip_runtime.h>

typedef __attribute__((ext_vector_type(16))) _Float16 v16h;
typedef __attribute__((ext_vector_type(8)))  float    v8f;

#define WMMA_F32_F16(a,b,c) \
  __builtin_amdgcn_wmma_f32_16x16x32_f16(false,(a),false,(b),(short)0,(c),false,false)

// ---- WMMA fragment placement helpers (ISA 7.12.2), one tile = 32 lanes x 16 halves = 512 halves
// A tile 16x32 f16: lane = m + 16*((k>>3)&1); elem = ((k>>4)<<3) | (k&7)
__device__ __forceinline__ int a_off(int m, int k) {
  int lane = (m & 15) | (((k >> 3) & 1) << 4);
  int elem = ((k >> 4) << 3) | (k & 7);
  return lane * 16 + elem;
}
// B tile 32x16 f16: lane = n + 16*((k>>3)&1); elem = ((k>>4)<<3) | (k&7)
__device__ __forceinline__ int b_off(int k, int n) {
  int lane = (n & 15) | (((k >> 3) & 1) << 4);
  int elem = ((k >> 4) << 3) | (k & 7);
  return lane * 16 + elem;
}
// Swin mask region for window coordinate wp (0..31) and local index (0..7)
__device__ __forceinline__ int mask_reg(int wp, int local) {
  return (wp == 31) ? (local < 4 ? 1 : 2) : 0;
}

// ---------------- weight swizzle kernels ----------------
// qkv 1x1 weights: (192,64) -> A tiles (12 mt x 2 kt)
__global__ void swz_qkv(const float* __restrict__ w, _Float16* __restrict__ dst) {
  int idx = blockIdx.x * 256 + threadIdx.x;              // 192*64 = 12288
  if (idx >= 192 * 64) return;
  int m = idx >> 6, k = idx & 63;
  int tile = (m >> 4) * 2 + (k >> 5);
  dst[tile * 512 + a_off(m & 15, k & 31)] = (_Float16)w[idx];
}
// proj 3x3 weights: OIHW (64,64,3,3) -> A tiles (4 mt x 18 kt), K = tap*64 + c
__global__ void swz_proj(const float* __restrict__ w, _Float16* __restrict__ dst) {
  int idx = blockIdx.x * 256 + threadIdx.x;              // 64*576 = 36864
  if (idx >= 64 * 576) return;
  int o = idx / 576, kk = idx % 576;
  int tap = kk >> 6, c = kk & 63;
  float v = w[(o * 64 + c) * 9 + tap];
  int tile = (o >> 4) * 18 + (kk >> 5);
  dst[tile * 512 + a_off(o & 15, kk & 31)] = (_Float16)v;
}

// ---------------- qkv 1x1 conv GEMM: M=192, K=64(pseudo-ch), N=128 (2 pseudo-windows) ----------------
template<typename TIN, bool SHIFT>
__global__ void __launch_bounds__(256)
gemm_qkv(const TIN* __restrict__ x, const _Float16* __restrict__ wA,
         _Float16* __restrict__ qkv) {
  __shared__ __align__(32) _Float16 ldsA[24 * 512];      // 12 mt x 2 kt  (24 KB)
  __shared__ __align__(32) _Float16 ldsB[16 * 512];      // 2 kt x 8 nt   (16 KB)
  const int tid = threadIdx.x;
  const int nw0 = blockIdx.x * 2;                        // first pseudo-window of this block

  for (int i = tid; i < 24 * 512; i += 256) ldsA[i] = wA[i];

  // stage activations: 64 pseudo-channels x 128 pixels, gather from image layout
  for (int i = tid; i < 64 * 128; i += 256) {
    int cp = i >> 7;                                     // pseudo-channel c'
    int p  = i & 127;
    int pw = p >> 6, pi = p & 63;
    int nq = nw0 + pw;
    int b = nq >> 10, rest = nq & 1023;
    int c = rest >> 4, hb = rest & 15;
    int h = hb * 16 + ((cp >> 5) << 3) + (pi >> 3);
    int w = ((cp & 31) << 3) + (pi & 7);
    if (SHIFT) { h = (h + 4) & 255; w = (w + 4) & 255; } // roll(-4,-4): read (h+4, w+4)
    float v = (float)x[((size_t)(b * 64 + c) << 16) + (h << 8) + w];
    int tile = (cp >> 5) * 8 + (p >> 4);
    ldsB[tile * 512 + b_off(cp & 31, p & 15)] = (_Float16)v;
  }
  __syncthreads();

  const int wv = tid >> 5, lane = tid & 31;              // wave = N-tile
  v8f acc[12] = {};
  v16h bfrag0 = *(const v16h*)&ldsB[(0 * 8 + wv) * 512 + lane * 16];
  v16h bfrag1 = *(const v16h*)&ldsB[(1 * 8 + wv) * 512 + lane * 16];
  #pragma unroll
  for (int mt = 0; mt < 12; ++mt) {
    v16h a0 = *(const v16h*)&ldsA[(mt * 2 + 0) * 512 + lane * 16];
    acc[mt] = WMMA_F32_F16(a0, bfrag0, acc[mt]);
    v16h a1 = *(const v16h*)&ldsA[(mt * 2 + 1) * 512 + lane * 16];
    acc[mt] = WMMA_F32_F16(a1, bfrag1, acc[mt]);
  }
  // store: qkv layout (8192, 192, 64) f16
  const int n = lane & 15, half = lane >> 4;
  const int p = wv * 16 + n;
  const size_t nq = (size_t)(nw0 + (p >> 6));
  const int pi = p & 63;
  #pragma unroll
  for (int mt = 0; mt < 12; ++mt)
    #pragma unroll
    for (int r = 0; r < 8; ++r) {
      int mch = mt * 16 + r + half * 8;
      qkv[(nq * 192 + mch) * 64 + pi] = (_Float16)acc[mt][r];
    }
}

// ---------------- fused depthwise 3x3 (patch-padded) ----------------
__device__ __forceinline__ void dw_patch(const _Float16* __restrict__ src,
                                         const float* __restrict__ wd9, float* o) {
  float in[64];
  #pragma unroll
  for (int i = 0; i < 64; ++i) in[i] = (float)src[i];
  #pragma unroll
  for (int r = 0; r < 8; ++r)
    #pragma unroll
    for (int s = 0; s < 8; ++s) {
      float a = 0.f;
      #pragma unroll
      for (int dy = -1; dy <= 1; ++dy)
        #pragma unroll
        for (int dx = -1; dx <= 1; ++dx) {
          int rr = r + dy, ss = s + dx;
          if (rr >= 0 && rr < 8 && ss >= 0 && ss < 8)
            a += wd9[(dy + 1) * 3 + (dx + 1)] * in[rr * 8 + ss];
        }
      o[r * 8 + s] = a;
    }
}

// ---------------- attention: 1 workgroup (128 thr / 4 waves) per pseudo-window ----------------
template<bool MASKED>
__global__ void __launch_bounds__(128)
win_attn(const _Float16* __restrict__ qkv, const float* __restrict__ wdw,
         _Float16* __restrict__ outp) {
  __shared__ __align__(32) _Float16 qA[8 * 512];         // A frags GEMM1: M=i, K=ch
  __shared__ __align__(32) _Float16 kB[8 * 512];         // B frags GEMM1: K=ch, N=j
  __shared__ __align__(32) _Float16 vA[8 * 512];         // A frags GEMM2: M=c', K=i
  __shared__ __align__(32) _Float16 pB[8 * 512];         // B frags GEMM2: K=i, N=j
  __shared__ float sAttn[64 * 64];
  const int tid = threadIdx.x;
  const size_t nq = blockIdx.x;
  const size_t base = nq * 192 * 64;

  { // q (rows 0..63) / k (rows 64..127): dwconv + l2norm + fragment store
    int ch = tid;
    float wd9[9], o[64];
    #pragma unroll
    for (int t = 0; t < 9; ++t) wd9[t] = wdw[ch * 9 + t];
    dw_patch(qkv + base + (size_t)ch * 64, wd9, o);
    float ss = 0.f;
    #pragma unroll
    for (int i = 0; i < 64; ++i) ss += o[i] * o[i];
    float scale = 1.f / fmaxf(sqrtf(ss), 1e-12f);
    int m = ch & 63, kt = m >> 5, kl = m & 31;
    if (ch < 64) {
      #pragma unroll
      for (int i = 0; i < 64; ++i)
        qA[((i >> 4) * 2 + kt) * 512 + a_off(i, kl)] = (_Float16)(o[i] * scale);
    } else {
      #pragma unroll
      for (int j = 0; j < 64; ++j)
        kB[(kt * 4 + (j >> 4)) * 512 + b_off(kl, j)] = (_Float16)(o[j] * scale);
    }
  }
  if (tid < 64) { // v (rows 128..191)
    int ch = 128 + tid;
    float wd9[9], o[64];
    #pragma unroll
    for (int t = 0; t < 9; ++t) wd9[t] = wdw[ch * 9 + t];
    dw_patch(qkv + base + (size_t)ch * 64, wd9, o);
    int mm = tid;
    #pragma unroll
    for (int i = 0; i < 64; ++i)
      vA[((mm >> 4) * 2 + (i >> 5)) * 512 + a_off(mm, i & 31)] = (_Float16)o[i];
  }
  __syncthreads();

  const int wv = tid >> 5, lane = tid & 31;
  { // GEMM1: attn = q^T k / 8  (wave wv owns M-tile wv, all 4 N-tiles)
    v8f a4[4] = {};
    #pragma unroll
    for (int kt = 0; kt < 2; ++kt) {
      v16h a = *(const v16h*)&qA[(wv * 2 + kt) * 512 + lane * 16];
      #pragma unroll
      for (int nt = 0; nt < 4; ++nt) {
        v16h b = *(const v16h*)&kB[(kt * 4 + nt) * 512 + lane * 16];
        a4[nt] = WMMA_F32_F16(a, b, a4[nt]);
      }
    }
    const int half = lane >> 4, n = lane & 15;
    int hnm = 0, wnm = 0;
    if (MASKED) { int widx = (int)nq & 1023; hnm = widx >> 5; wnm = widx & 31; }
    #pragma unroll
    for (int nt = 0; nt < 4; ++nt)
      #pragma unroll
      for (int r = 0; r < 8; ++r) {
        int i = wv * 16 + r + half * 8;
        int j = nt * 16 + n;
        float v = a4[nt][r] * 0.125f;
        if (MASKED) {
          int ri = mask_reg(hnm, i >> 3) * 3 + mask_reg(wnm, i & 7);
          int rj = mask_reg(hnm, j >> 3) * 3 + mask_reg(wnm, j & 7);
          if (ri != rj) v -= 100.f;
        }
        sAttn[i * 64 + j] = v;
      }
  }
  __syncthreads();

  if (tid < 64) { // softmax over j for row i = tid, write B frags for GEMM2
    const float* row = &sAttn[tid * 64];
    float mx = row[0];
    #pragma unroll
    for (int j = 1; j < 64; ++j) mx = fmaxf(mx, row[j]);
    float e[64], s = 0.f;
    #pragma unroll
    for (int j = 0; j < 64; ++j) { e[j] = __expf(row[j] - mx); s += e[j]; }
    float inv = 1.f / s;
    int i = tid, kt = i >> 5, kl = i & 31;
    #pragma unroll
    for (int j = 0; j < 64; ++j)
      pB[(kt * 4 + (j >> 4)) * 512 + b_off(kl, j)] = (_Float16)(e[j] * inv);
  }
  __syncthreads();

  { // GEMM2: out = v x attn
    v8f a4[4] = {};
    #pragma unroll
    for (int kt = 0; kt < 2; ++kt) {
      v16h a = *(const v16h*)&vA[(wv * 2 + kt) * 512 + lane * 16];
      #pragma unroll
      for (int nt = 0; nt < 4; ++nt) {
        v16h b = *(const v16h*)&pB[(kt * 4 + nt) * 512 + lane * 16];
        a4[nt] = WMMA_F32_F16(a, b, a4[nt]);
      }
    }
    const int half = lane >> 4, n = lane & 15;
    #pragma unroll
    for (int nt = 0; nt < 4; ++nt)
      #pragma unroll
      for (int r = 0; r < 8; ++r) {
        int cp = wv * 16 + r + half * 8;
        int j = nt * 16 + n;
        outp[(nq * 64 + cp) * 64 + j] = (_Float16)a4[nt][r];   // (8192,64,64) pseudo layout
      }
  }
}

// ---------------- proj 3x3 conv (patch-padded im2col GEMM): 4 windows per workgroup ----------------
template<bool FINAL>
__global__ void __launch_bounds__(256)
proj_conv(const _Float16* __restrict__ pin, const _Float16* __restrict__ wA,
          _Float16* __restrict__ out16, float* __restrict__ out32) {
  __shared__ __align__(32) _Float16 ldsA[72 * 512];      // 4 mt x 18 kt (72 KB)
  __shared__ __align__(32) _Float16 ldsB[32 * 512];      // 2 kt x 16 nt (32 KB)
  const int tid = threadIdx.x;
  const int w0 = blockIdx.x * 4;
  for (int i = tid; i < 72 * 512; i += 256) ldsA[i] = wA[i];

  const int wv = tid >> 5, lane = tid & 31;
  v8f acc[2][4] = {};

  for (int tap = 0; tap < 9; ++tap) {
    const int dy = tap / 3 - 1, dx = tap % 3 - 1;
    __syncthreads();
    // stage im2col B chunk: 64 c' x 256 pixels (4 windows), zero-pad inside 8x8 patch
    for (int i = tid; i < 64 * 256; i += 256) {
      int cp = i >> 8;
      int p  = i & 255;
      int win = p >> 6, pi = p & 63;
      int r = (pi >> 3) + dy, s = (pi & 7) + dx;
      float v = 0.f;
      if (r >= 0 && r < 8 && s >= 0 && s < 8)
        v = (float)pin[(((size_t)(w0 + win)) * 64 + cp) * 64 + r * 8 + s];
      int tile = (cp >> 5) * 16 + (p >> 4);
      ldsB[tile * 512 + b_off(cp & 31, p & 15)] = (_Float16)v;
    }
    __syncthreads();
    #pragma unroll
    for (int sub = 0; sub < 2; ++sub) {
      int nt = wv * 2 + sub;
      #pragma unroll
      for (int ktl = 0; ktl < 2; ++ktl) {
        v16h b = *(const v16h*)&ldsB[(ktl * 16 + nt) * 512 + lane * 16];
        int ktg = tap * 2 + ktl;
        #pragma unroll
        for (int mt = 0; mt < 4; ++mt) {
          v16h a = *(const v16h*)&ldsA[(mt * 18 + ktg) * 512 + lane * 16];
          acc[sub][mt] = WMMA_F32_F16(a, b, acc[sub][mt]);
        }
      }
    }
  }
  // window-reverse store back to image layout
  const int half = lane >> 4, n = lane & 15;
  #pragma unroll
  for (int sub = 0; sub < 2; ++sub) {
    int nt = wv * 2 + sub;
    int nq = w0 + (nt >> 2);
    int b = nq >> 10, rest = nq & 1023, c = rest >> 4, hb = rest & 15;
    int j = (nt & 3) * 16 + n;
    int pr = j >> 3, ps = j & 7;
    #pragma unroll
    for (int mt = 0; mt < 4; ++mt)
      #pragma unroll
      for (int r = 0; r < 8; ++r) {
        int cp = mt * 16 + r + half * 8;
        int h = hb * 16 + ((cp >> 5) << 3) + pr;
        int w = ((cp & 31) << 3) + ps;
        if (FINAL) {       // final un-roll(+4,+4): value computed at (h,w) lands at (h+4,w+4)
          int hf = (h + 4) & 255, wf = (w + 4) & 255;
          out32[(((size_t)b * 64 + c) << 16) + (hf << 8) + wf] = acc[sub][mt][r];
        } else {
          out16[(((size_t)b * 64 + c) << 16) + (h << 8) + w] = (_Float16)acc[sub][mt][r];
        }
      }
  }
}

// ---------------- host launch ----------------
extern "C" void kernel_launch(void* const* d_in, const int* in_sizes, int n_in,
                              void* d_out, int out_size, void* d_ws, size_t ws_size,
                              hipStream_t stream) {
  const float* x       = (const float*)d_in[0];
  const float* w_qkv0  = (const float*)d_in[1];
  const float* w_dw0   = (const float*)d_in[2];
  const float* w_proj0 = (const float*)d_in[3];
  const float* w_qkv1  = (const float*)d_in[4];
  const float* w_dw1   = (const float*)d_in[5];
  const float* w_proj1 = (const float*)d_in[6];

  // workspace layout (~336 MB total):
  //   qkv  : (8192,192,64) f16  = 201326592 B
  //   mid1 : (8192, 64,64) f16  =  67108864 B   (attention output, pseudo layout)
  //   mid2 : (8,64,256,256) f16 =  67108864 B   (pass-1 image-layout output)
  //   swizzled weights          =    196608 B
  char* ws = (char*)d_ws;
  _Float16* qkv  = (_Float16*)(ws);
  _Float16* mid1 = (_Float16*)(ws + 201326592u);
  _Float16* mid2 = (_Float16*)(ws + 268435456u);
  _Float16* wA0  = (_Float16*)(ws + 335544320u);
  _Float16* wA1  = wA0 + 12288;
  _Float16* wP0  = wA1 + 12288;
  _Float16* wP1  = wP0 + 36864;

  swz_qkv <<<48, 256, 0, stream>>>(w_qkv0, wA0);
  swz_qkv <<<48, 256, 0, stream>>>(w_qkv1, wA1);
  swz_proj<<<144, 256, 0, stream>>>(w_proj0, wP0);
  swz_proj<<<144, 256, 0, stream>>>(w_proj1, wP1);

  // pass 1: unshifted
  gemm_qkv<float, false><<<4096, 256, 0, stream>>>(x, wA0, qkv);
  win_attn<false>       <<<8192, 128, 0, stream>>>(qkv, w_dw0, mid1);
  proj_conv<false>      <<<2048, 256, 0, stream>>>(mid1, wP0, mid2, nullptr);

  // pass 2: shifted (-4,-4) with Swin mask, final un-shift folded into store
  gemm_qkv<_Float16, true><<<4096, 256, 0, stream>>>(mid2, wA1, qkv);
  win_attn<true>          <<<8192, 128, 0, stream>>>(qkv, w_dw1, mid1);
  proj_conv<true>         <<<2048, 256, 0, stream>>>(mid1, wP1, nullptr, (float*)d_out);
}
